// RecurrentSkipLayer_1666447311270
// MI455X (gfx1250) — matrix operand order repaired and checked by reading, not compiled, not run on hardware
//
#include <hip/hip_runtime.h>
#include <hip/hip_bf16.h>

// ---------------------------------------------------------------------------
// Problem constants (from reference): B=64, S=2048, I=128, H=256, P=24
// ---------------------------------------------------------------------------
#define B_SZ   64
#define S_SZ   2048
#define I_SZ   128
#define H_SZ   256
#define H3     768
#define PSKIP  24

typedef __bf16 bf16;
typedef bf16           v16bf __attribute__((ext_vector_type(16)));
typedef float          v8f   __attribute__((ext_vector_type(8)));
typedef float          v4f   __attribute__((ext_vector_type(4)));
typedef unsigned short u16x8 __attribute__((ext_vector_type(8)));
typedef unsigned short u16x4 __attribute__((ext_vector_type(4)));

union Frag {
    v16bf          v;
    u16x8          h[2];
    unsigned short us[16];
};

__device__ __forceinline__ unsigned short f2bf_bits(float f) {
    unsigned u = __float_as_uint(f);
    unsigned r = u + 0x7FFFu + ((u >> 16) & 1u);   // round-to-nearest-even
    return (unsigned short)(r >> 16);
}

__device__ __forceinline__ v8f wmma_bf16(v16bf a, v16bf b, v8f c) {
    return __builtin_amdgcn_wmma_f32_16x16x32_bf16(
        /*neg_a=*/false, a, /*neg_b=*/false, b,
        /*c_mod=*/(short)0, c, /*reuse_a=*/false, /*reuse_b=*/false);
}

// B-matrix fragment (32x16, K x N). Lane L: n = nrow (caller folds lane&15 in),
// k = k0 + (L>>4)*16 + {0..15} contiguous -> native row-major weight row chunk.
__device__ __forceinline__ v16bf load_b_frag(const unsigned short* W,
                                             int nrow, int ldk, int k0, int hi) {
    const unsigned short* p = W + (size_t)nrow * ldk + k0 + (hi << 4);
    Frag f;
    f.h[0] = *(const u16x8*)(p);
    f.h[1] = *(const u16x8*)(p + 8);
    return f.v;
}

// A-matrix fragment (16x32) from a bf16 row in global memory.
// Lane L: m = L&15 (caller passes its row pointer), halves 0..15 contiguous
// at k0+hi*16.
__device__ __forceinline__ v16bf load_a_bf(const unsigned short* row,
                                           int k0, int hi) {
    const unsigned short* p = row + k0 + hi * 16;
    Frag f;
    f.h[0] = *(const u16x8*)(p);
    f.h[1] = *(const u16x8*)(p + 8);
    return f.v;
}

// ---------------------------------------------------------------------------
// Kernel 1a: fp32 -> bf16 weight conversion (weights then live in L2)
// ---------------------------------------------------------------------------
__global__ void convert_weights(const float* __restrict__ Wih,
                                const float* __restrict__ Whh,
                                const float* __restrict__ Wsk,
                                const float* __restrict__ Wout,
                                unsigned short* __restrict__ o_ih,
                                unsigned short* __restrict__ o_hh,
                                unsigned short* __restrict__ o_sk,
                                unsigned short* __restrict__ o_out) {
    const int n_ih  = H3 * I_SZ;
    const int n_hh  = H3 * H_SZ;
    const int n_sk  = H_SZ * H_SZ;
    const int n_out = I_SZ * H_SZ;
    const int total = n_ih + n_hh + n_sk + n_out;
    for (int i = blockIdx.x * blockDim.x + threadIdx.x; i < total;
         i += gridDim.x * blockDim.x) {
        if (i < n_ih)                 o_ih[i] = f2bf_bits(Wih[i]);
        else if (i < n_ih + n_hh)     o_hh[i - n_ih] = f2bf_bits(Whh[i - n_ih]);
        else if (i < n_ih + n_hh + n_sk)
            o_sk[i - n_ih - n_hh] = f2bf_bits(Wsk[i - n_ih - n_hh]);
        else
            o_out[i - n_ih - n_hh - n_sk] = f2bf_bits(Wout[i - n_ih - n_hh - n_sk]);
    }
}

// ---------------------------------------------------------------------------
// Kernel 1b: fp32 -> bf16 conversion of x (off the sequential critical path;
// removes all conversion VALU from the scan loop). Vectorized 4-wide.
// ---------------------------------------------------------------------------
__global__ void convert_x(const float* __restrict__ x,
                          unsigned short* __restrict__ xb) {
    const size_t total4 = (size_t)B_SZ * S_SZ * I_SZ / 4;
    for (size_t j = blockIdx.x * (size_t)blockDim.x + threadIdx.x; j < total4;
         j += (size_t)gridDim.x * blockDim.x) {
        v4f f = *(const v4f*)(x + 4 * j);
        u16x4 o;
#pragma unroll
        for (int k = 0; k < 4; ++k) o[k] = f2bf_bits(f[k]);
        *(u16x4*)(xb + 4 * j) = o;
    }
}

// ---------------------------------------------------------------------------
// Kernel 2: sequential GRU scan. 4 workgroups x 16 batch rows, 16 waves each.
// Wave w owns hidden columns [16w, 16w+16): it computes the r/z/n/skip tiles
// for those columns, so the full gate math for an element stays in one lane.
// Weight fragments are re-read from L2 every step (asm memory clobber blocks
// LICM so they are NOT hoisted into registers -> no scratch spills).
// ---------------------------------------------------------------------------
__global__ __launch_bounds__(512) void gru_scan(
    const unsigned short* __restrict__ xb,  // [B,S,I] bf16
    const unsigned short* __restrict__ Wih, // [768,128] bf16
    const unsigned short* __restrict__ Whh, // [768,256] bf16
    const unsigned short* __restrict__ Wsk, // [256,256] bf16
    const float* __restrict__ b_ih,         // [768]
    const float* __restrict__ b_hh,         // [768]
    const float* __restrict__ b_skip,       // [256]
    unsigned short* __restrict__ hsb)       // [S,B,H] bf16 hidden history
{
    __shared__ unsigned short lds_h[16 * H_SZ];   // h_{t-1} for this batch slice

    const int lane = threadIdx.x & 31;
    const int wave = threadIdx.x >> 5;            // 0..15
    const int hi   = lane >> 4;
    const int m    = lane & 15;                   // A-row / D-col lane index
    const int b0   = blockIdx.x * 16;             // batch slice
    const int nl   = wave * 16 + m;               // owned hidden column

    // Biases (r and z gates fold gi+gh biases together; n gate cannot).
    const float bias_r  = b_ih[nl] + b_hh[nl];
    const float bias_z  = b_ih[H_SZ + nl] + b_hh[H_SZ + nl];
    const float bias_in = b_ih[2 * H_SZ + nl];
    const float bias_hn = b_hh[2 * H_SZ + nl];
    const float bias_sk = b_skip[nl];

    for (int i = threadIdx.x; i < 16 * H_SZ; i += blockDim.x) lds_h[i] = 0;
    v8f hprev;
#pragma unroll
    for (int r = 0; r < 8; ++r) hprev[r] = 0.0f;
    __syncthreads();

    const unsigned short* xrow = xb + (size_t)(b0 + m) * S_SZ * I_SZ;

#pragma unroll 1
    for (int t = 0; t < S_SZ; ++t) {
        // Block LICM: without this the compiler hoists ~44 loop-invariant
        // weight fragments (~350 VGPRs) out of the loop and spills to scratch.
        asm volatile("" ::: "memory");

        v8f accR, accZ, accIN, accHN, accSK;
#pragma unroll
        for (int r = 0; r < 8; ++r) {
            accR[r] = bias_r;  accZ[r] = bias_z;
            accIN[r] = bias_in; accHN[r] = bias_hn; accSK[r] = bias_sk;
        }

        // gi contribution: x_t[16,128] @ W_ih^T  (K = 128, 4 WMMA steps)
        const unsigned short* xt = xrow + (size_t)t * I_SZ;
#pragma unroll
        for (int kk = 0; kk < I_SZ / 32; ++kk) {
            const int k0 = kk * 32;
            v16bf a = load_a_bf(xt, k0, hi);
            accR  = wmma_bf16(a, load_b_frag(Wih, nl,            I_SZ, k0, hi), accR);
            accZ  = wmma_bf16(a, load_b_frag(Wih, H_SZ + nl,     I_SZ, k0, hi), accZ);
            accIN = wmma_bf16(a, load_b_frag(Wih, 2 * H_SZ + nl, I_SZ, k0, hi), accIN);
        }

        // gh contribution: h[16,256] @ W_hh^T  (K = 256, 8 WMMA steps), A from LDS
#pragma unroll
        for (int kk = 0; kk < H_SZ / 32; ++kk) {
            const int k0 = kk * 32;
            Frag fa;
            fa.h[0] = *(const u16x8*)(&lds_h[m * H_SZ + k0 + hi * 16]);
            fa.h[1] = *(const u16x8*)(&lds_h[m * H_SZ + k0 + hi * 16 + 8]);
            accR  = wmma_bf16(fa.v, load_b_frag(Whh, nl,            H_SZ, k0, hi), accR);
            accZ  = wmma_bf16(fa.v, load_b_frag(Whh, H_SZ + nl,     H_SZ, k0, hi), accZ);
            accHN = wmma_bf16(fa.v, load_b_frag(Whh, 2 * H_SZ + nl, H_SZ, k0, hi), accHN);
        }

        // skip contribution: h_{t-P}[16,256] @ W_skip^T  (uniform branch)
        const bool do_skip = (t >= PSKIP);
        if (do_skip) {
            const unsigned short* srow =
                hsb + ((size_t)(t - PSKIP) * B_SZ + b0 + m) * H_SZ;
#pragma unroll
            for (int kk = 0; kk < H_SZ / 32; ++kk) {
                const int k0 = kk * 32;
                v16bf a = load_a_bf(srow, k0, hi);
                accSK = wmma_bf16(a, load_b_frag(Wsk, nl, H_SZ, k0, hi), accSK);
            }
        }
        __syncthreads();   // all waves finished reading lds_h (h_{t-1})

        // Gate math: D-fragment element (mm, nl) lives in this lane, reg r.
        v8f hnew;
#pragma unroll
        for (int r = 0; r < 8; ++r) {
            float rg = 1.0f / (1.0f + __expf(-accR[r]));
            float zg = 1.0f / (1.0f + __expf(-accZ[r]));
            float ng = tanhf(accIN[r] + rg * accHN[r]);
            float hv = (1.0f - zg) * ng + zg * hprev[r];
            if (do_skip) hv += accSK[r];
            hnew[r] = hv;
        }
        hprev = hnew;

        // Publish h_t: bf16 to LDS (next step's A) and global history buffer.
        const int mbase = hi * 8;
        unsigned short* grow = hsb + ((size_t)t * B_SZ + b0) * H_SZ;
#pragma unroll
        for (int r = 0; r < 8; ++r) {
            const int mm = mbase + r;
            const unsigned short hb = f2bf_bits(hnew[r]);
            lds_h[mm * H_SZ + nl]        = hb;
            grow[(size_t)mm * H_SZ + nl] = hb;
        }
        __syncthreads();   // h_t visible to all waves before next iteration
    }
}

// ---------------------------------------------------------------------------
// Kernel 3: output projection  out[b,i,t] = sum_k hs[t,b,k]*W_out[i,k] + b_out[i]
// One 16x16 (i x t) tile per wave; 65536 tiles total, fully parallel.
// ---------------------------------------------------------------------------
__global__ __launch_bounds__(256) void out_proj(
    const unsigned short* __restrict__ hsb,   // [S,B,H] bf16
    const unsigned short* __restrict__ Wout,  // [128,256] bf16
    const float* __restrict__ b_out,          // [128]
    float* __restrict__ out)                  // [B,I,S] fp32
{
    const int lane = threadIdx.x & 31;
    const int wave = threadIdx.x >> 5;
    const int tile = blockIdx.x * (blockDim.x >> 5) + wave;   // 0..65535
    const int tt = tile & 127;          // S/16 = 128
    const int it = (tile >> 7) & 7;     // I/16 = 8
    const int b  = tile >> 10;          // B    = 64
    const int i0 = it * 16, t0 = tt * 16;
    const int m  = lane & 15, hi = lane >> 4;

    v8f acc;
    const float* bp = b_out + i0 + hi * 8;
#pragma unroll
    for (int r = 0; r < 8; ++r) acc[r] = bp[r];   // bias depends on row i

    const unsigned short* arow = Wout + (size_t)(i0 + m) * H_SZ;          // A: W_out rows
    const unsigned short* brow = hsb + ((size_t)(t0 + m) * B_SZ + b) * H_SZ + hi * 16; // B: hs^T
#pragma unroll
    for (int kk = 0; kk < H_SZ / 32; ++kk) {
        const int k0 = kk * 32;
        Frag fa, fb;
        fa.h[0] = *(const u16x8*)(arow + k0 + hi * 16);
        fa.h[1] = *(const u16x8*)(arow + k0 + hi * 16 + 8);
        fb.h[0] = *(const u16x8*)(brow + k0);
        fb.h[1] = *(const u16x8*)(brow + k0 + 8);
        acc = wmma_bf16(fa.v, fb.v, acc);
    }

    // Store out[b][i0 + hi*8 + r][t0 + m]
    float* op = out + ((size_t)b * I_SZ + i0 + hi * 8) * S_SZ + t0 + m;
#pragma unroll
    for (int r = 0; r < 8; ++r) op[(size_t)r * S_SZ] = acc[r];
}

// ---------------------------------------------------------------------------
// Host launcher
// ---------------------------------------------------------------------------
extern "C" void kernel_launch(void* const* d_in, const int* in_sizes, int n_in,
                              void* d_out, int out_size, void* d_ws, size_t ws_size,
                              hipStream_t stream) {
    (void)in_sizes; (void)n_in; (void)out_size; (void)ws_size;
    const float* x     = (const float*)d_in[0];
    const float* W_ih  = (const float*)d_in[1];
    const float* W_hh  = (const float*)d_in[2];
    const float* b_ih  = (const float*)d_in[3];
    const float* b_hh  = (const float*)d_in[4];
    const float* W_sk  = (const float*)d_in[5];
    const float* b_sk  = (const float*)d_in[6];
    const float* W_out = (const float*)d_in[7];
    const float* b_out = (const float*)d_in[8];
    // d_in[9] = skip_period (24) baked in as PSKIP.

    char* ws = (char*)d_ws;
    unsigned short* wih  = (unsigned short*)(ws);                            // 768*128*2
    unsigned short* whh  = (unsigned short*)(ws + 196608);                   // 768*256*2
    unsigned short* wsk  = (unsigned short*)(ws + 196608 + 393216);          // 256*256*2
    unsigned short* wout = (unsigned short*)(ws + 196608 + 393216 + 131072); // 128*256*2
    unsigned short* xbuf = (unsigned short*)(ws + 786432);                   // [B,S,I] bf16
    unsigned short* hsb  = (unsigned short*)(ws + 786432 + 33554432);        // [S,B,H] bf16

    convert_weights<<<384, 256, 0, stream>>>(W_ih, W_hh, W_sk, W_out,
                                             wih, whh, wsk, wout);
    convert_x<<<4096, 256, 0, stream>>>(x, xbuf);
    gru_scan<<<B_SZ / 16, 512, 0, stream>>>(xbuf, wih, whh, wsk,
                                            b_ih, b_hh, b_sk, hsb);
    out_proj<<<(B_SZ * (I_SZ / 16) * (S_SZ / 16)) / 8, 256, 0, stream>>>(
        hsb, wout, b_out, (float*)d_out);
}
